// RNN_Encoder_26405458936010
// MI455X (gfx1250) — matrix operand order repaired and checked by reading, not compiled, not run on hardware
//
#include <hip/hip_runtime.h>
#include <hip/hip_bf16.h>

// ---------------------------------------------------------------------------
// AWD-LSTM encoder for MI455X (gfx1250, wave32, WMMA).
//
// Pipeline per layer l (din -> H):
//   1) convert W_ih / W_hh to bf16 with per-gate padded rows  [4*HpN, Kp]
//   2) bias[g*HpN+n] = b_ih + b_hh (f32)
//   3) XG = X_l @ W_ih^T + bias     (big parallel WMMA GEMM, f32 out)
//   4) persistent scan kernel: 512 sequential steps, each
//        g = XG[t] + h @ W_hh^T  (4 gate WMMA tiles per wave)
//        c = sig(f)*c + sig(i)*tanh(g); h = sig(o)*tanh(c)
//      hidden state double-buffered in bf16 through L2, one grid barrier/step;
//      each block stages its 16 A-rows in LDS per step (blocks are mt-uniform)
//      so the post-barrier L2 re-reads happen once per block, not per wave.
// Matrix core: v_wmma_f32_16x16x32_bf16, K padded to 32, N per-gate padded to 16.
// ---------------------------------------------------------------------------

typedef __bf16 bf16_t;
typedef __attribute__((ext_vector_type(8)))  bf16_t v8bf;
typedef __attribute__((ext_vector_type(16))) bf16_t v16bf;
typedef __attribute__((ext_vector_type(8)))  float  v8f;

#define LDS_PAD 8  // row stride pad: (1152+8)*2B = 580 dwords == 4 mod 64 banks

// A-matrix (16x32 bf16) per-lane fragment: lanes 0-15 hold K {0..7,16..23},
// lanes 16-31 hold K {8..15,24..31}; two contiguous 16B chunks per lane.
__device__ __forceinline__ v16bf load_a_frag(const bf16_t* row, int k0, int khalf) {
  v8bf lo = *(const v8bf*)(row + k0 + khalf * 8);
  v8bf hi = *(const v8bf*)(row + k0 + 16 + khalf * 8);
  return __builtin_shufflevector(lo, hi, 0,1,2,3,4,5,6,7,8,9,10,11,12,13,14,15);
}
// B-matrix (32x16 bf16) per-lane fragment: lane holds column n = lane&15,
// same K order as A: lanes 0-15 K {0..7,16..23}, lanes 16-31 K {8..15,24..31};
// two 16B chunks per lane (row-major
// weight row, since B = W^T means B's column n is W's row n).
__device__ __forceinline__ v16bf load_b_frag(const bf16_t* __restrict__ row, int k0, int khalf) {
  v8bf lo = *(const v8bf*)(row + k0 + khalf * 8);
  v8bf hi = *(const v8bf*)(row + k0 + 16 + khalf * 8);
  return __builtin_shufflevector(lo, hi, 0,1,2,3,4,5,6,7,8,9,10,11,12,13,14,15);
}

__device__ __forceinline__ float sigmoidf_(float x) { return 1.0f / (1.0f + __expf(-x)); }

// Device-scope grid barrier (persistent kernel; grid is <= 36 blocks).
__device__ __forceinline__ void grid_sync(unsigned* cnt, unsigned* gen, unsigned nblocks) {
  __threadfence();
  __syncthreads();
  if (threadIdx.x == 0) {
    unsigned g = __hip_atomic_load(gen, __ATOMIC_ACQUIRE, __HIP_MEMORY_SCOPE_AGENT);
    unsigned arrived =
        __hip_atomic_fetch_add(cnt, 1u, __ATOMIC_ACQ_REL, __HIP_MEMORY_SCOPE_AGENT) + 1u;
    if (arrived == nblocks) {
      __hip_atomic_fetch_add(cnt, (unsigned)(0u - nblocks), __ATOMIC_ACQ_REL,
                             __HIP_MEMORY_SCOPE_AGENT);
      __hip_atomic_fetch_add(gen, 1u, __ATOMIC_ACQ_REL, __HIP_MEMORY_SCOPE_AGENT);
    } else {
      while (__hip_atomic_load(gen, __ATOMIC_ACQUIRE, __HIP_MEMORY_SCOPE_AGENT) == g)
        __builtin_amdgcn_s_sleep(1);
    }
  }
  __syncthreads();
}

// ---- embedding lookup + f32->bf16, zero K-padding -------------------------
__global__ __launch_bounds__(256) void embed_kernel(const int* __restrict__ x,
                                                    const float* __restrict__ emb,
                                                    bf16_t* __restrict__ X0,
                                                    int Kp, int K, int total) {
  int i = blockIdx.x * blockDim.x + threadIdx.x;
  if (i >= total) return;
  int k = i % Kp, m = i / Kp;
  float v = 0.0f;
  if (k < K) v = emb[(size_t)x[m] * K + k];
  X0[i] = (bf16_t)v;
}

// ---- weight f32 [4H,K] -> bf16 per-gate padded [4*HpN, Kp] ----------------
__global__ __launch_bounds__(256) void wconvert_kernel(const float* __restrict__ src,
                                                       bf16_t* __restrict__ dst,
                                                       int H, int HpN, int K, int Kp, int total) {
  int i = blockIdx.x * blockDim.x + threadIdx.x;
  if (i >= total) return;
  int k = i % Kp, r = i / Kp;
  int n = r % HpN, g = r / HpN;
  float v = 0.0f;
  if (n < H && k < K) v = src[((size_t)g * H + n) * K + k];
  dst[i] = (bf16_t)v;
}

__global__ __launch_bounds__(256) void bias_kernel(const float* __restrict__ bih,
                                                   const float* __restrict__ bhh,
                                                   float* __restrict__ dst,
                                                   int H, int HpN, int total) {
  int i = blockIdx.x * blockDim.x + threadIdx.x;
  if (i >= total) return;
  int n = i % HpN, g = i / HpN;
  dst[i] = (n < H) ? (bih[g * H + n] + bhh[g * H + n]) : 0.0f;
}

// ---- XG = X @ W_ih^T + bias (bf16 in, f32 out), grid-stride WMMA GEMM -----
__global__ __launch_bounds__(256) void gemm_xg_kernel(const bf16_t* __restrict__ X,
                                                      const bf16_t* __restrict__ W,
                                                      const float* __restrict__ bias,
                                                      float* __restrict__ XG,
                                                      int Kp, int HpN, int Gp, int ksteps,
                                                      int Mtiles, int NnT) {
  const int lane  = threadIdx.x & 31;
  const int lrow  = lane & 15;
  const int khalf = lane >> 4;
  const int nwaves = (gridDim.x * blockDim.x) >> 5;
  const int wid    = (blockIdx.x * blockDim.x + threadIdx.x) >> 5;
  const int ntasks = Mtiles * NnT;

  for (int task = wid; task < ntasks; task += nwaves) {
    const int mt = task / NnT, nt = task % NnT;  // waves in a block share mt -> A reuse in WGP$
    const int m0 = mt << 4, n0 = nt << 4;
    const bf16_t* arow = X + (size_t)(m0 + lrow) * Kp;

    v8f acc[4];
#pragma unroll
    for (int g = 0; g < 4; ++g) {
      float bv = bias[g * HpN + n0 + lrow];  // column value, broadcast over rows
#pragma unroll
      for (int v = 0; v < 8; ++v) acc[g][v] = bv;
    }

    for (int kb = 0; kb < ksteps; ++kb) {
      const int k0 = kb * 32;
      v16bf av = load_a_frag(arow, k0, khalf);
#pragma unroll
      for (int g = 0; g < 4; ++g) {
        const bf16_t* brow = W + ((size_t)g * HpN + n0 + lrow) * Kp;
        v16bf bv = load_b_frag(brow, k0, khalf);
        acc[g] = __builtin_amdgcn_wmma_f32_16x16x32_bf16(false, av, false, bv,
                                                         (short)0, acc[g], false, false);
      }
    }
    // D layout: element v of lane -> row m0+v+8*khalf, col n0+(lane&15)
#pragma unroll
    for (int g = 0; g < 4; ++g) {
      float* orow = XG + (size_t)(m0 + 8 * khalf) * Gp + (size_t)g * HpN + n0 + lrow;
#pragma unroll
      for (int v = 0; v < 8; ++v) orow[(size_t)v * Gp] = acc[g][v];
    }
  }
}

// ---- persistent LSTM scan: 512 sequential steps, grid barrier per step ----
// Grid: 4 * blocksPerMt blocks of 256 threads; block's mt = blockIdx/blocksPerMt
// so the whole block shares one 16-row A tile, staged in LDS each step.
__global__ __launch_bounds__(256) void lstm_scan_kernel(
    const float* __restrict__ XG, const bf16_t* __restrict__ Whh,
    bf16_t* __restrict__ hb0, bf16_t* __restrict__ hb1, float* __restrict__ cbuf,
    bf16_t* __restrict__ Xnext, int ldxn, float* __restrict__ Yout,
    int S, int H, int HpN, int Khp, int Gp, int ksteps, int NnT, int blocksPerMt,
    unsigned* cnt, unsigned* gen, unsigned nblocks) {
  __shared__ __align__(16) bf16_t smem[16 * (1152 + LDS_PAD)];  // 37,120 B static LDS

  const int lane  = threadIdx.x & 31;
  const int lrow  = lane & 15;
  const int khalf = lane >> 4;
  const int mt = blockIdx.x / blocksPerMt;                       // uniform per block
  const int nt = (blockIdx.x % blocksPerMt) * 8 + (threadIdx.x >> 5);
  const bool active = nt < NnT;

  const int m0 = mt << 4;
  const int n0 = nt << 4;
  const int ncol = n0 + lrow;          // D column for this lane
  const int ldsStride = Khp + LDS_PAD;
  const int chunksPerRow = Khp >> 3;   // 16B chunks per A row
  const int totalChunks = chunksPerRow << 4;

  const bf16_t* brow[4];
#pragma unroll
  for (int g = 0; g < 4; ++g)
    brow[g] = Whh + ((size_t)g * HpN + n0 + lrow) * Khp;
  const bf16_t* arow_lds = smem + lrow * ldsStride;

  for (int t = 0; t < S; ++t) {
    const bf16_t* hcur = (t & 1) ? hb1 : hb0;
    bf16_t*       hnxt = (t & 1) ? hb0 : hb1;

    // cooperative stage of this block's 16 A rows: global (L2) -> LDS, once
    for (int c = threadIdx.x; c < totalChunks; c += 256) {
      int r = c / chunksPerRow, cc = c - r * chunksPerRow;
      *(v8bf*)(smem + r * ldsStride + (cc << 3)) =
          *(const v8bf*)(hcur + (size_t)(m0 + r) * Khp + (cc << 3));
    }
    __syncthreads();

    if (active) {
      const float* xgbase =
          XG + ((size_t)t * 64 + m0 + 8 * khalf) * Gp + n0 + lrow;
      // pull next step's XG rows toward the caches while we compute this step
      if (t + 1 < S) __builtin_prefetch(xgbase + (size_t)64 * Gp, 0, 1);

      v8f acc[4];
#pragma unroll
      for (int g = 0; g < 4; ++g)
#pragma unroll
        for (int v = 0; v < 8; ++v)
          acc[g][v] = xgbase[(size_t)v * Gp + (size_t)g * HpN];

      for (int kb = 0; kb < ksteps; ++kb) {
        const int k0 = kb * 32;
        v16bf av = load_a_frag(arow_lds, k0, khalf);   // ds_load_b128 x2
#pragma unroll
        for (int g = 0; g < 4; ++g) {
          v16bf bv = load_b_frag(brow[g], k0, khalf);  // global_load_b128 x2 (L2 stream)
          acc[g] = __builtin_amdgcn_wmma_f32_16x16x32_bf16(false, av, false, bv,
                                                           (short)0, acc[g], false, false);
        }
      }

      // LSTM cell update, fully in-register (each (m,n) owned by one wave)
#pragma unroll
      for (int v = 0; v < 8; ++v) {
        const int m = m0 + v + 8 * khalf;
        float ig = sigmoidf_(acc[0][v]);
        float fg = sigmoidf_(acc[1][v]);
        float gg = tanhf(acc[2][v]);
        float og = sigmoidf_(acc[3][v]);
        float c_old = cbuf[(size_t)m * HpN + ncol];
        float c_new = fg * c_old + ig * gg;
        cbuf[(size_t)m * HpN + ncol] = c_new;
        float h = og * tanhf(c_new);
        if (ncol < H) {
          hnxt[(size_t)m * Khp + ncol] = (bf16_t)h;
          if (Xnext) Xnext[((size_t)t * 64 + m) * ldxn + ncol] = (bf16_t)h;
          if (Yout)  Yout[((size_t)t * 64 + m) * H + ncol] = h;
        }
      }
    }
    // writes of step t visible (and LDS free for re-stage) before step t+1
    grid_sync(cnt, gen, nblocks);
  }
}

// ---------------------------------------------------------------------------
extern "C" void kernel_launch(void* const* d_in, const int* in_sizes, int n_in,
                              void* d_out, int out_size, void* d_ws, size_t ws_size,
                              hipStream_t stream) {
  (void)in_sizes; (void)n_in; (void)out_size; (void)ws_size;
  const int S = 512, B = 64, M = S * B;
  (void)B;

  const int*   x   = (const int*)d_in[0];
  const float* emb = (const float*)d_in[1];
  struct Lp { const float *wih, *whh, *bih, *bhh; int din, H; };
  Lp L[3] = {
    {(const float*)d_in[2],  (const float*)d_in[3],  (const float*)d_in[4],  (const float*)d_in[5],  400, 1150},
    {(const float*)d_in[6],  (const float*)d_in[7],  (const float*)d_in[8],  (const float*)d_in[9],  1150, 1150},
    {(const float*)d_in[10], (const float*)d_in[11], (const float*)d_in[12], (const float*)d_in[13], 1150, 400},
  };
  const int KinP[3] = {416, 1152, 1152};  // input K padded to 32
  const int HpN[3]  = {1152, 1152, 400};  // per-gate N pad to 16
  const int Khp[3]  = {1152, 1152, 416};  // H as recurrent K, padded to 32
  const int Gp[3]   = {4608, 4608, 1600}; // 4*HpN

  // workspace bump allocator (256B aligned)
  char* ws = (char*)d_ws;
  size_t off = 0;
  auto alloc = [&](size_t bytes) -> void* {
    void* p = ws + off;
    off += (bytes + 255) & ~(size_t)255;
    return p;
  };
  bf16_t* X0 = (bf16_t*)alloc((size_t)M * 416 * 2);
  bf16_t* X1 = (bf16_t*)alloc((size_t)M * 1152 * 2);
  bf16_t* X2 = (bf16_t*)alloc((size_t)M * 1152 * 2);
  float*  XG = (float*)alloc((size_t)M * 4608 * 4);
  bf16_t* Wih[3]; bf16_t* Whh[3]; float* bias[3];
  for (int l = 0; l < 3; ++l) {
    Wih[l]  = (bf16_t*)alloc((size_t)4 * HpN[l] * KinP[l] * 2);
    Whh[l]  = (bf16_t*)alloc((size_t)4 * HpN[l] * Khp[l] * 2);
    bias[l] = (float*)alloc((size_t)4 * HpN[l] * 4);
  }
  bf16_t* hb0 = (bf16_t*)alloc((size_t)64 * 1152 * 2);
  bf16_t* hb1 = (bf16_t*)alloc((size_t)64 * 1152 * 2);
  float*  cb  = (float*)alloc((size_t)64 * 1152 * 4);
  unsigned* ctr = (unsigned*)alloc(256);

  // zero the inter-layer activation buffers once (keeps K-padding columns 0)
  hipMemsetAsync(X1, 0, (size_t)M * 1152 * 2, stream);
  hipMemsetAsync(X2, 0, (size_t)M * 1152 * 2, stream);

  // embedding -> bf16 X0
  {
    int total = M * 416;
    embed_kernel<<<(total + 255) / 256, 256, 0, stream>>>(x, emb, X0, 416, 400, total);
  }

  bf16_t* Xin[3]  = {X0, X1, X2};
  bf16_t* Xout[3] = {X1, X2, nullptr};

  for (int l = 0; l < 3; ++l) {
    // weights / bias to padded bf16 layout
    {
      int tw = 4 * HpN[l] * KinP[l];
      wconvert_kernel<<<(tw + 255) / 256, 256, 0, stream>>>(L[l].wih, Wih[l], L[l].H, HpN[l],
                                                            L[l].din, KinP[l], tw);
    }
    {
      int tw = 4 * HpN[l] * Khp[l];
      wconvert_kernel<<<(tw + 255) / 256, 256, 0, stream>>>(L[l].whh, Whh[l], L[l].H, HpN[l],
                                                            L[l].H, Khp[l], tw);
    }
    {
      int tb = 4 * HpN[l];
      bias_kernel<<<(tb + 255) / 256, 256, 0, stream>>>(L[l].bih, L[l].bhh, bias[l],
                                                        L[l].H, HpN[l], tb);
    }
    // batched input GEMM: XG = X @ W_ih^T + bias
    {
      int Mtiles = M / 16, NnT = HpN[l] / 16;
      gemm_xg_kernel<<<1024, 256, 0, stream>>>(Xin[l], Wih[l], bias[l], XG,
                                               KinP[l], HpN[l], Gp[l], KinP[l] / 32,
                                               Mtiles, NnT);
    }
    // recurrent scan (persistent): init state, then 512 steps in one kernel
    hipMemsetAsync(hb0, 0, (size_t)64 * 1152 * 2, stream);
    hipMemsetAsync(hb1, 0, (size_t)64 * 1152 * 2, stream);
    hipMemsetAsync(cb,  0, (size_t)64 * 1152 * 4, stream);
    hipMemsetAsync(ctr, 0, 256, stream);
    {
      int NnT = HpN[l] / 16;
      int blocksPerMt = (NnT + 7) / 8;      // 8 waves (one N-tile each) per block
      int nblocks = 4 * blocksPerMt;        // 4 M-tiles (B=64), mt uniform per block
      lstm_scan_kernel<<<nblocks, 256, 0, stream>>>(
          XG, Whh[l], hb0, hb1, cb, Xout[l], 1152,
          (l == 2) ? (float*)d_out : nullptr,
          S, L[l].H, HpN[l], Khp[l], Gp[l], Khp[l] / 32, NnT, blocksPerMt,
          ctr, ctr + 1, (unsigned)nblocks);
    }
  }
}
